// get_model_9620726743423
// MI455X (gfx1250) — compile-verified, matrix operand order checked by reading, not run on hardware
//
#include <hip/hip_runtime.h>
#include <hip/hip_bf16.h>

// ---------------------------------------------------------------------------
// VecKM + Point-Transformer forward for MI455X (gfx1250, wave32, WMMA).
// GEMMs: 128x64 block tile, 8 waves, 4 x v_wmma_f32_16x16x32_f16 per wave per
// K-step. LDS tiles staged in f32 via GLOBAL_LOAD_ASYNC_TO_LDS_B128 (ASYNCcnt
// tracked, s_wait_asynccnt) with a plain b128 copy fallback; f32->f16
// conversion fused into the fragment build. Channels-last [B,N,C] layout.
// ---------------------------------------------------------------------------

typedef __attribute__((ext_vector_type(16))) _Float16 v16h;
typedef __attribute__((ext_vector_type(8)))  float    v8f;

#define TM     128   // block tile rows
#define TN     64    // block tile cols
#define TK     32    // K step
#define LDS_W  36    // LDS row stride (f32 elems); 36*4=144B keeps 16B alignment

#if defined(__HIP_DEVICE_COMPILE__) && \
    __has_builtin(__builtin_amdgcn_global_load_async_to_lds_b128) && \
    __has_builtin(__builtin_amdgcn_s_wait_asynccnt)
#define HAVE_ASYNC_LDS 1
#else
#define HAVE_ASYNC_LDS 0
#endif

#if HAVE_ASYNC_LDS
typedef int async_v4i __attribute__((vector_size(16)));
#define ASYNC_COPY16(gp, lp) \
    __builtin_amdgcn_global_load_async_to_lds_b128((async_v4i*)(gp), (async_v4i*)(lp), 0, 0)
#endif

// ---------------------------------------------------------------------------
// Generic batched GEMM: C = alpha * op(A) * op(B) + beta * C
//   TA==0: A element (m,k) = A[m*lda + k]    (row-major [M,K])
//   TA==1: A element (m,k) = A[k*lda + m]    (A stored [K,M])
//   TB==0: B given as Bt [N,K] row-major (weights): (k,n) = B[n*ldb+k]
//   TB==1: B given as    [K,N] row-major:           (k,n) = B[k*ldb+n]
// ---------------------------------------------------------------------------
template<int TA, int TB>
__global__ __launch_bounds__(256) void gemm_wmma_kernel(
    const float* __restrict__ A, const float* __restrict__ B, float* __restrict__ C,
    int M, int N, int K, int lda, int ldb, int ldc,
    long long sA, long long sB, long long sC,
    float alpha, float beta)
{
    __shared__ float sa[TM * LDS_W];   // A tile [row][k]
    __shared__ float sb[TN * LDS_W];   // B tile [col][k]

    const int bz = blockIdx.z;
    A += (long long)bz * sA;
    B += (long long)bz * sB;
    C += (long long)bz * sC;

    const int mBlock = blockIdx.y * TM;
    const int nBlock = blockIdx.x * TN;
    const int tid  = threadIdx.x;
    const int lane = tid & 31;
    const int wave = tid >> 5;          // 8 waves; wave = 16-row strip

    const bool fullM = (mBlock + TM <= M);
    const bool fullN = (nBlock + TN <= N);
    const bool alnA  = ((lda & 3) == 0);
    const bool alnB  = ((ldb & 3) == 0);

    v8f acc[4] = {};

    for (int k0 = 0; k0 < K; k0 += TK) {
        const bool fullK = (k0 + TK <= K);
        bool asyncUsed = false;

        // ---------------- stage A tile ----------------
        if (TA == 0 && fullM && fullK && alnA) {
            // contiguous in k: 1024 x 16B transfers, 4 per thread
#pragma unroll
            for (int i = 0; i < (TM * TK / 4) / 256; ++i) {
                int v = tid + i * 256;
                int r = v >> 3;             // 8 vec4 per row
                int kc = (v & 7) * 4;
                const float* gp = A + (long long)(mBlock + r) * lda + (k0 + kc);
                float* lp = &sa[r * LDS_W + kc];
#if HAVE_ASYNC_LDS
                ASYNC_COPY16(gp, lp);
#else
                *(float4*)lp = *(const float4*)gp;
#endif
            }
#if HAVE_ASYNC_LDS
            asyncUsed = true;
#endif
        } else if (TA == 1 && fullM && fullK && alnA) {
            // contiguous in m: vectorized load, scatter into LDS rows
#pragma unroll
            for (int i = 0; i < (TM * TK / 4) / 256; ++i) {
                int v = tid + i * 256;
                int k  = v >> 5;            // TM/4 = 32 vec4 per k
                int mc = (v & 31) * 4;
                const float4 f = *(const float4*)(A + (long long)(k0 + k) * lda + (mBlock + mc));
                sa[(mc + 0) * LDS_W + k] = f.x;
                sa[(mc + 1) * LDS_W + k] = f.y;
                sa[(mc + 2) * LDS_W + k] = f.z;
                sa[(mc + 3) * LDS_W + k] = f.w;
            }
        } else {
            for (int t = tid; t < TM * TK; t += 256) {
                int r  = t >> 5;
                int kk = t & 31;
                int gm = mBlock + r, gk = k0 + kk;
                float v = 0.f;
                if (gm < M && gk < K)
                    v = TA ? A[(long long)gk * lda + gm] : A[(long long)gm * lda + gk];
                sa[r * LDS_W + kk] = v;
            }
        }

        // ---------------- stage B tile ----------------
        if (TB == 0 && fullN && fullK && alnB) {
#pragma unroll
            for (int i = 0; i < (TN * TK / 4) / 256; ++i) {
                int v = tid + i * 256;
                int c  = v >> 3;
                int kc = (v & 7) * 4;
                const float* gp = B + (long long)(nBlock + c) * ldb + (k0 + kc);
                float* lp = &sb[c * LDS_W + kc];
#if HAVE_ASYNC_LDS
                ASYNC_COPY16(gp, lp);
#else
                *(float4*)lp = *(const float4*)gp;
#endif
            }
#if HAVE_ASYNC_LDS
            asyncUsed = true;
#endif
        } else if (TB == 1 && fullN && fullK && alnB) {
#pragma unroll
            for (int i = 0; i < (TN * TK / 4) / 256; ++i) {
                int v = tid + i * 256;
                int k  = v >> 4;            // TN/4 = 16 vec4 per k
                int nc = (v & 15) * 4;
                const float4 f = *(const float4*)(B + (long long)(k0 + k) * ldb + (nBlock + nc));
                sb[(nc + 0) * LDS_W + k] = f.x;
                sb[(nc + 1) * LDS_W + k] = f.y;
                sb[(nc + 2) * LDS_W + k] = f.z;
                sb[(nc + 3) * LDS_W + k] = f.w;
            }
        } else {
            for (int t = tid; t < TN * TK; t += 256) {
                int c  = t >> 5;
                int kk = t & 31;
                int gn = nBlock + c, gk = k0 + kk;
                float v = 0.f;
                if (gn < N && gk < K)
                    v = TB ? B[(long long)gk * ldb + gn] : B[(long long)gn * ldb + gk];
                sb[c * LDS_W + kk] = v;
            }
        }

#if HAVE_ASYNC_LDS
        if (asyncUsed) __builtin_amdgcn_s_wait_asynccnt(0);
#else
        (void)asyncUsed;
#endif
        __syncthreads();

        // ---------------- fragments + WMMA ----------------
        // f16 A-matrix 16x32 layout: lanes 0-15 hold K {0..7,16..23},
        // lanes 16-31 hold K {8..15,24..31}; pairs per VGPR.
        const int hlf = lane >> 4;
        const int rw  = lane & 15;
        const float* pa = &sa[(wave * 16 + rw) * LDS_W];
        v16h fa;
#pragma unroll
        for (int i = 0; i < 8; ++i) {
            int kk = ((i & 4) ? 16 : 0) + hlf * 8 + (i & 3) * 2;
            fa[2 * i]     = (_Float16)pa[kk];
            fa[2 * i + 1] = (_Float16)pa[kk + 1];
        }
#pragma unroll
        for (int j = 0; j < 4; ++j) {
            const float* pb = &sb[(j * 16 + rw) * LDS_W];
            v16h fb;
#pragma unroll
            for (int i = 0; i < 8; ++i) {
                int kk = ((i & 4) ? 16 : 0) + hlf * 8 + (i & 3) * 2;
                fb[2 * i]     = (_Float16)pb[kk];
                fb[2 * i + 1] = (_Float16)pb[kk + 1];
            }
            acc[j] = __builtin_amdgcn_wmma_f32_16x16x32_f16(false, fa, false, fb,
                                                            (short)0, acc[j], false, false);
        }
        __syncthreads();
    }

    // ---------------- store ----------------
    // C/D layout: col = base + (lane&15); VGPR r -> row r + 8*(lane>>4)
    const int hlf = lane >> 4;
    const int cw  = lane & 15;
    const int mBase = mBlock + wave * 16;
#pragma unroll
    for (int j = 0; j < 4; ++j) {
        int col = nBlock + j * 16 + cw;
        if (col < N) {
#pragma unroll
            for (int r = 0; r < 8; ++r) {
                int rowG = mBase + r + 8 * hlf;
                if (rowG < M) {
                    long long idx = (long long)rowG * ldc + col;
                    float prev = (beta != 0.f) ? C[idx] : 0.f;
                    C[idx] = alpha * acc[j][r] + beta * prev;
                }
            }
        }
    }
}

// ---------------------------------------------------------------------------
// Elementwise / reduction helper kernels
// ---------------------------------------------------------------------------

// x = act( (x + pre_bias?) * (gamma/sqrt(1+eps))? + beta? ); act:0 none,1 relu,2 leaky0.2
__global__ void rowwise_affine_act_kernel(float* __restrict__ x,
    const float* __restrict__ pre_bias, const float* __restrict__ gamma,
    const float* __restrict__ beta, long long total, int C, int act)
{
    long long i = (long long)blockIdx.x * blockDim.x + threadIdx.x;
    if (i >= total) return;
    int c = (int)(i % C);
    float v = x[i];
    if (pre_bias) v += pre_bias[c];
    if (gamma)    v *= gamma[c] * rsqrtf(1.f + 1e-5f);
    if (beta)     v += beta[c];
    if (act == 1)      v = fmaxf(v, 0.f);
    else if (act == 2) v = (v > 0.f) ? v : 0.2f * v;
    x[i] = v;
}

__global__ void sub_kernel(const float* __restrict__ a, const float* __restrict__ b,
                           float* __restrict__ out, long long total)
{
    long long i = (long long)blockIdx.x * blockDim.x + threadIdx.x;
    if (i >= total) return;
    out[i] = a[i] - b[i];
}

// Y = X + relu( (T + tb) * scale + bb )
__global__ void sa_epilogue_kernel(const float* __restrict__ X, const float* __restrict__ T,
    const float* __restrict__ tb, const float* __restrict__ g, const float* __restrict__ bb,
    float* __restrict__ Y, long long total, int C)
{
    long long i = (long long)blockIdx.x * blockDim.x + threadIdx.x;
    if (i >= total) return;
    int c = (int)(i % C);
    float v = T[i] + tb[c];
    v = v * (g[c] * rsqrtf(1.f + 1e-5f)) + bb[c];
    v = fmaxf(v, 0.f);
    Y[i] = X[i] + v;
}

// out = (R + br)^2 + (I + bi)^2
__global__ void complex_bias_sq_kernel(const float* __restrict__ R, const float* __restrict__ I,
    const float* __restrict__ br, const float* __restrict__ bi,
    float* __restrict__ out, long long total, int C)
{
    long long i = (long long)blockIdx.x * blockDim.x + threadIdx.x;
    if (i >= total) return;
    int c = (int)(i % C);
    float r  = R[i] + br[c];
    float im = I[i] + bi[c];
    out[i] = r * r + im * im;
}

// eA = exp(i * pts@A); also per-point squared norm. block = 256 (=d), grid = B*N
__global__ void veckm_pre_kernel(const float* __restrict__ x, const float* __restrict__ A,
    float* __restrict__ eAr, float* __restrict__ eAi, float* __restrict__ sq, int N, int d)
{
    int bn = blockIdx.x;
    int c  = threadIdx.x;
    int b  = bn / N, n = bn % N;
    const float px = x[(long long)b * 3 * N + n];
    const float py = x[(long long)b * 3 * N + N + n];
    const float pz = x[(long long)b * 3 * N + 2 * N + n];
    float ang = px * A[c] + py * A[d + c] + pz * A[2 * d + c];
    long long o = (long long)bn * d + c;
    eAr[o] = cosf(ang);
    eAi[o] = sinf(ang);
    if (c == 0) sq[bn] = px * px + py * py + pz * pz;
}

// J[b,n,m] = exp(-18 * ||p_n - p_m||^2)
__global__ void jmat_kernel(const float* __restrict__ x, const float* __restrict__ sq,
                            float* __restrict__ J, int N)
{
    int m = blockIdx.x * blockDim.x + threadIdx.x;
    int n = blockIdx.y;
    int b = blockIdx.z;
    if (m >= N) return;
    const float* xb = x + (long long)b * 3 * N;
    float dot = xb[n] * xb[m] + xb[N + n] * xb[N + m] + xb[2 * N + n] * xb[2 * N + m];
    float d2  = sq[b * N + n] + sq[b * N + m] - 2.f * dot;
    J[((long long)b * N + n) * N + m] = expf(-18.f * d2);
}

// G = (Jc)/eA (|eA|=1 -> multiply by conjugate), row-normalize *16, add exp(i pts@P)
__global__ void veckm_finish_kernel(const float* __restrict__ x, const float* __restrict__ P,
    const float* __restrict__ eAr, const float* __restrict__ eAi,
    float* __restrict__ Gr, float* __restrict__ Gi, int N, int d)
{
    __shared__ float red[256];
    int bn = blockIdx.x;
    int c  = threadIdx.x;
    int b  = bn / N, n = bn % N;
    long long o = (long long)bn * d + c;
    float gr = Gr[o], gi = Gi[o];
    float cr = eAr[o], ci = eAi[o];
    float g1r = gr * cr + gi * ci;
    float g1i = gi * cr - gr * ci;
    red[c] = g1r * g1r + g1i * g1i;
    __syncthreads();
    for (int s = 128; s > 0; s >>= 1) { if (c < s) red[c] += red[c + s]; __syncthreads(); }
    float scale = 16.f * rsqrtf(red[0] + 1e-30f);
    const float px = x[(long long)b * 3 * N + n];
    const float py = x[(long long)b * 3 * N + N + n];
    const float pz = x[(long long)b * 3 * N + 2 * N + n];
    float ang = px * P[c] + py * P[d + c] + pz * P[2 * d + c];
    Gr[o] = g1r * scale + cosf(ang);
    Gi[o] = g1i * scale + sinf(ang);
}

// row softmax over last axis (row length N=1024), block 256, 4 elems/thread
__global__ void softmax_row_kernel(float* __restrict__ att, int N)
{
    __shared__ float red[256];
    float* row = att + (long long)blockIdx.x * N;
    int t = threadIdx.x;
    float v[4];
    float mx = -3.0e38f;
#pragma unroll
    for (int j = 0; j < 4; ++j) { v[j] = row[t + j * 256]; mx = fmaxf(mx, v[j]); }
    red[t] = mx; __syncthreads();
    for (int s = 128; s > 0; s >>= 1) { if (t < s) red[t] = fmaxf(red[t], red[t + s]); __syncthreads(); }
    mx = red[0]; __syncthreads();
    float sum = 0.f;
#pragma unroll
    for (int j = 0; j < 4; ++j) { v[j] = expf(v[j] - mx); sum += v[j]; }
    red[t] = sum; __syncthreads();
    for (int s = 128; s > 0; s >>= 1) { if (t < s) red[t] += red[t + s]; __syncthreads(); }
    float inv = 1.f / red[0];
#pragma unroll
    for (int j = 0; j < 4; ++j) row[t + j * 256] = v[j] * inv;
}

// cs[b,m] = sum_n att[b,n,m]
__global__ void colsum_kernel(const float* __restrict__ att, float* __restrict__ cs, int N)
{
    int m = blockIdx.x * 256 + threadIdx.x;
    int b = blockIdx.y;
    if (m >= N) return;
    const float* base = att + (long long)b * N * N;
    float s = 0.f;
    for (int n = 0; n < N; ++n) s += base[(long long)n * N + m];
    cs[b * N + m] = s;
}

__global__ void colnorm_kernel(float* __restrict__ att, const float* __restrict__ cs,
                               int N, long long total)
{
    long long i = (long long)blockIdx.x * blockDim.x + threadIdx.x;
    if (i >= total) return;
    int m = (int)(i % N);
    int b = (int)(i / ((long long)N * N));
    att[i] /= (1e-9f + cs[b * N + m]);
}

// cat[b,n,0:1024]=f0..f3, [1024:1280]=GC  (all sources [B,N,256])
__global__ void concat_kernel(const float* __restrict__ f0, const float* __restrict__ f1,
    const float* __restrict__ f2, const float* __restrict__ f3, const float* __restrict__ gc,
    float* __restrict__ cat, long long total)
{
    long long i = (long long)blockIdx.x * blockDim.x + threadIdx.x;
    if (i >= total) return;
    int c = (int)(i % 1280);
    long long bn = i / 1280;
    const float* src; int cc;
    if      (c < 256)  { src = f0; cc = c; }
    else if (c < 512)  { src = f1; cc = c - 256; }
    else if (c < 768)  { src = f2; cc = c - 512; }
    else if (c < 1024) { src = f3; cc = c - 768; }
    else               { src = gc; cc = c - 1024; }
    cat[i] = src[bn * 256 + cc];
}

// out[b,c] = max_n h[b,n,c]
__global__ void maxpool_kernel(const float* __restrict__ h, float* __restrict__ out,
                               int B, int N, int C)
{
    int i = blockIdx.x * 256 + threadIdx.x;
    if (i >= B * C) return;
    int b = i / C, c = i % C;
    const float* base = h + (long long)b * N * C + c;
    float mx = -3.0e38f;
    for (int n = 0; n < N; ++n) mx = fmaxf(mx, base[(long long)n * C]);
    out[i] = mx;
}

// ---------------------------------------------------------------------------
// Host-side orchestration
// ---------------------------------------------------------------------------

static inline void launch_gemm(hipStream_t st, int ta, int tb,
    const float* A, const float* B, float* C,
    int M, int N, int K, int lda, int ldb, int ldc,
    long long sA, long long sB, long long sC, int batch,
    float alpha, float beta)
{
    dim3 grid((N + TN - 1) / TN, (M + TM - 1) / TM, batch);
    dim3 block(256);
    if (ta == 0 && tb == 0)
        gemm_wmma_kernel<0, 0><<<grid, block, 0, st>>>(A, B, C, M, N, K, lda, ldb, ldc, sA, sB, sC, alpha, beta);
    else if (ta == 0 && tb == 1)
        gemm_wmma_kernel<0, 1><<<grid, block, 0, st>>>(A, B, C, M, N, K, lda, ldb, ldc, sA, sB, sC, alpha, beta);
    else
        gemm_wmma_kernel<1, 1><<<grid, block, 0, st>>>(A, B, C, M, N, K, lda, ldb, ldc, sA, sB, sC, alpha, beta);
}

static inline long long cdivll(long long a, long long b) { return (a + b - 1) / b; }

extern "C" void kernel_launch(void* const* d_in, const int* in_sizes, int n_in,
                              void* d_out, int out_size, void* d_ws, size_t ws_size,
                              hipStream_t stream)
{
    (void)in_sizes; (void)n_in; (void)out_size; (void)ws_size;

    const float* x       = (const float*)d_in[0];
    const float* Amat    = (const float*)d_in[1];
    const float* Pmat    = (const float*)d_in[2];
    const float* cl1_wr  = (const float*)d_in[3];
    const float* cl1_wi  = (const float*)d_in[4];
    const float* cl1_br  = (const float*)d_in[5];
    const float* cl1_bi  = (const float*)d_in[6];
    const float* cl2_wr  = (const float*)d_in[7];
    const float* cl2_wi  = (const float*)d_in[8];
    const float* cl2_br  = (const float*)d_in[9];
    const float* cl2_bi  = (const float*)d_in[10];
    const float* conv1_w = (const float*)d_in[11];
    const float* conv2_w = (const float*)d_in[12];
    const float* bn1_g   = (const float*)d_in[13];
    const float* bn1_b   = (const float*)d_in[14];
    const float* bn2_g   = (const float*)d_in[15];
    const float* bn2_b   = (const float*)d_in[16];
    const float* qk_w    = (const float*)d_in[17];
    const float* v_w     = (const float*)d_in[18];
    const float* v_b     = (const float*)d_in[19];
    const float* t_w     = (const float*)d_in[20];
    const float* t_b     = (const float*)d_in[21];
    const float* sabn_g  = (const float*)d_in[22];
    const float* sabn_b  = (const float*)d_in[23];
    const float* fuse_w  = (const float*)d_in[24];
    const float* fbn_g   = (const float*)d_in[25];
    const float* fbn_b   = (const float*)d_in[26];
    const float* lin1_w  = (const float*)d_in[27];
    const float* bn6_g   = (const float*)d_in[28];
    const float* bn6_b   = (const float*)d_in[29];
    const float* lin2_w  = (const float*)d_in[30];
    const float* lin2_b  = (const float*)d_in[31];
    const float* bn7_g   = (const float*)d_in[32];
    const float* bn7_b   = (const float*)d_in[33];
    const float* lin3_w  = (const float*)d_in[34];
    const float* lin3_b  = (const float*)d_in[35];

    const int B = 16, N = 1024, d = 256;
    const long long BNtot = (long long)B * N;              // 16384 rows
    const long long TE = BNtot * d;                        // elems per [B,N,256]

    // -------- workspace arena (MiB offsets; peak ~282 MiB) --------
    char* w = (char*)d_ws;
    const size_t MiB = 1ull << 20;
    auto F = [&](size_t mb) -> float* { return (float*)(w + mb * MiB); };
    float* eAr = F(0);            // [B,N,256]
    float* eAi = F(16);
    float* Gr  = F(32);           // J@eAr -> finished G.real; later CL2.real
    float* Gi  = F(48);
    float* Jm  = F(64);           // [B,N,N]  (also attention matrix)
    float* T1  = F(128);          // scratch [B,N,256]
    float* T2  = F(144);
    float* T3  = F(160);
    float* GC  = F(176);          // |G|^2 features (persist to concat)
    float* H0  = F(192);          // conv2 output
    float* f0  = F(208);          // SA outputs (feats)
    float* f1  = F(224);
    float* f2  = F(240);
    float* f3  = F(256);
    float* sq     = F(272);       // [B,N]
    float* xq     = F(273);       // [B,N,64]
    float* colsum = F(278);       // [B,N]
    float* pooled = F(279);       // [B,1024]
    float* fc1    = F(280);       // [16,512]
    float* fc2    = F(281);       // [16,256]
    float* cat      = F(64);      // 80 MiB, reuses att+T1 region (dead by then)
    float* fuse_out = F(144);     // 64 MiB, reuses T2,T3,GC,H0 region (dead by then)

    float* out = (float*)d_out;   // [16,40] f32

    // ---- 1) eA = exp(i pts@A), point squared norms ----
    veckm_pre_kernel<<<B * N, 256, 0, stream>>>(x, Amat, eAr, eAi, sq, N, d);

    // ---- 2) J = exp(-18 * pairwise dist^2) ----
    jmat_kernel<<<dim3(N / 256, N, B), 256, 0, stream>>>(x, sq, Jm, N);

    // ---- 3) G = J @ eA  (batched, M=1024, K=1024, cols=256) ----
    launch_gemm(stream, 0, 1, Jm, eAr, Gr, N, d, N, N, d, d,
                (long long)N * N, (long long)N * d, (long long)N * d, B, 1.f, 0.f);
    launch_gemm(stream, 0, 1, Jm, eAi, Gi, N, d, N, N, d, d,
                (long long)N * N, (long long)N * d, (long long)N * d, B, 1.f, 0.f);

    // ---- 4) divide by eA, normalize *16, add positional encoding ----
    veckm_finish_kernel<<<B * N, 256, 0, stream>>>(x, Pmat, eAr, eAi, Gr, Gi, N, d);

    const int EW = 256;
    // ---- 5) ComplexLinear1 + CReLU ----
    launch_gemm(stream, 0, 0, Gr, cl1_wr, T1, (int)BNtot, d, d, d, d, d, 0, 0, 0, 1,  1.f, 0.f);
    launch_gemm(stream, 0, 0, Gi, cl1_wi, T1, (int)BNtot, d, d, d, d, d, 0, 0, 0, 1, -1.f, 1.f);
    launch_gemm(stream, 0, 0, Gr, cl1_wi, T2, (int)BNtot, d, d, d, d, d, 0, 0, 0, 1,  1.f, 0.f);
    launch_gemm(stream, 0, 0, Gi, cl1_wr, T2, (int)BNtot, d, d, d, d, d, 0, 0, 0, 1,  1.f, 1.f);
    rowwise_affine_act_kernel<<<cdivll(TE, EW), EW, 0, stream>>>(T1, cl1_br, nullptr, nullptr, TE, d, 1);
    rowwise_affine_act_kernel<<<cdivll(TE, EW), EW, 0, stream>>>(T2, cl1_bi, nullptr, nullptr, TE, d, 1);

    // ---- 6) ComplexLinear2, then |.|^2 with bias ----
    launch_gemm(stream, 0, 0, T1, cl2_wr, Gr, (int)BNtot, d, d, d, d, d, 0, 0, 0, 1,  1.f, 0.f);
    launch_gemm(stream, 0, 0, T2, cl2_wi, Gr, (int)BNtot, d, d, d, d, d, 0, 0, 0, 1, -1.f, 1.f);
    launch_gemm(stream, 0, 0, T1, cl2_wi, Gi, (int)BNtot, d, d, d, d, d, 0, 0, 0, 1,  1.f, 0.f);
    launch_gemm(stream, 0, 0, T2, cl2_wr, Gi, (int)BNtot, d, d, d, d, d, 0, 0, 0, 1,  1.f, 1.f);
    complex_bias_sq_kernel<<<cdivll(TE, EW), EW, 0, stream>>>(Gr, Gi, cl2_br, cl2_bi, GC, TE, d);

    // ---- 7) conv1 -> bn relu -> conv2 -> bn relu ----
    launch_gemm(stream, 0, 0, GC, conv1_w, T1, (int)BNtot, 256, 256, 256, 256, 256, 0, 0, 0, 1, 1.f, 0.f);
    rowwise_affine_act_kernel<<<cdivll(TE, EW), EW, 0, stream>>>(T1, nullptr, bn1_g, bn1_b, TE, 256, 1);
    launch_gemm(stream, 0, 0, T1, conv2_w, H0, (int)BNtot, 256, 256, 256, 256, 256, 0, 0, 0, 1, 1.f, 0.f);
    rowwise_affine_act_kernel<<<cdivll(TE, EW), EW, 0, stream>>>(H0, nullptr, bn2_g, bn2_b, TE, 256, 1);

    // ---- 8) 4 self-attention layers ----
    float* att = Jm;
    float* feats[4] = { f0, f1, f2, f3 };
    const float* X = H0;
    const long long attTotal = (long long)B * N * N;
    for (int i = 0; i < 4; ++i) {
        const float* qk = qk_w + (long long)i * 64 * 256;
        const float* vw = v_w  + (long long)i * 256 * 256;
        const float* vb = v_b  + (long long)i * 256;
        const float* tw = t_w  + (long long)i * 256 * 256;
        const float* tb = t_b  + (long long)i * 256;
        const float* sg = sabn_g + (long long)i * 256;
        const float* sb = sabn_b + (long long)i * 256;

        // xq = X @ qk^T  -> [B,N,64]
        launch_gemm(stream, 0, 0, X, qk, xq, (int)BNtot, 64, 256, 256, 256, 64, 0, 0, 0, 1, 1.f, 0.f);
        // energy = xq @ xq^T per batch
        launch_gemm(stream, 0, 0, xq, xq, att, N, N, 64, 64, 64, N,
                    (long long)N * 64, (long long)N * 64, (long long)N * N, B, 1.f, 0.f);
        // softmax over last axis, then column-sum normalize
        softmax_row_kernel<<<B * N, 256, 0, stream>>>(att, N);
        colsum_kernel<<<dim3(N / 256, B), 256, 0, stream>>>(att, colsum, N);
        colnorm_kernel<<<cdivll(attTotal, EW), EW, 0, stream>>>(att, colsum, N, attTotal);
        // xv = X @ vw^T + vb
        launch_gemm(stream, 0, 0, X, vw, T1, (int)BNtot, 256, 256, 256, 256, 256, 0, 0, 0, 1, 1.f, 0.f);
        rowwise_affine_act_kernel<<<cdivll(TE, EW), EW, 0, stream>>>(T1, vb, nullptr, nullptr, TE, 256, 0);
        // xr = att^T @ xv per batch   (TA=1: att stored [n,m])
        launch_gemm(stream, 1, 1, att, T1, T2, N, 256, N, N, 256, 256,
                    (long long)N * N, (long long)N * 256, (long long)N * 256, B, 1.f, 0.f);
        // t = (X - xr) @ tw^T ; Y = X + relu(bn(t + tb))
        sub_kernel<<<cdivll(TE, EW), EW, 0, stream>>>(X, T2, T3, TE);
        launch_gemm(stream, 0, 0, T3, tw, T1, (int)BNtot, 256, 256, 256, 256, 256, 0, 0, 0, 1, 1.f, 0.f);
        sa_epilogue_kernel<<<cdivll(TE, EW), EW, 0, stream>>>(X, T1, tb, sg, sb, feats[i], TE, 256);
        X = feats[i];
    }

    // ---- 9) concat [f0..f3, GC] -> [B,N,1280] ----
    const long long catTotal = BNtot * 1280;
    concat_kernel<<<cdivll(catTotal, EW), EW, 0, stream>>>(f0, f1, f2, f3, GC, cat, catTotal);

    // ---- 10) fuse conv 1280->1024, bn + leaky(0.2) ----
    launch_gemm(stream, 0, 0, cat, fuse_w, fuse_out, (int)BNtot, 1024, 1280, 1280, 1280, 1024,
                0, 0, 0, 1, 1.f, 0.f);
    rowwise_affine_act_kernel<<<cdivll(BNtot * 1024, EW), EW, 0, stream>>>(fuse_out, nullptr, fbn_g, fbn_b, BNtot * 1024, 1024, 2);

    // ---- 11) global max pool over N ----
    maxpool_kernel<<<(B * 1024 + 255) / 256, 256, 0, stream>>>(fuse_out, pooled, B, N, 1024);

    // ---- 12) FC head ----
    launch_gemm(stream, 0, 0, pooled, lin1_w, fc1, B, 512, 1024, 1024, 1024, 512, 0, 0, 0, 1, 1.f, 0.f);
    rowwise_affine_act_kernel<<<cdivll((long long)B * 512, EW), EW, 0, stream>>>(fc1, nullptr, bn6_g, bn6_b, (long long)B * 512, 512, 2);
    launch_gemm(stream, 0, 0, fc1, lin2_w, fc2, B, 256, 512, 512, 512, 256, 0, 0, 0, 1, 1.f, 0.f);
    rowwise_affine_act_kernel<<<cdivll((long long)B * 256, EW), EW, 0, stream>>>(fc2, lin2_b, bn7_g, bn7_b, (long long)B * 256, 256, 2);
    launch_gemm(stream, 0, 0, fc2, lin3_w, out, B, 40, 256, 256, 256, 40, 0, 0, 0, 1, 1.f, 0.f);
    rowwise_affine_act_kernel<<<cdivll((long long)B * 40, EW), EW, 0, stream>>>(out, lin3_b, nullptr, nullptr, (long long)B * 40, 40, 0);
}
